// MambaBlock_13434657702441
// MI455X (gfx1250) — compile-verified, matrix operand order checked by reading, not run on hardware
//
#include <hip/hip_runtime.h>

#define DD 1024
#define BB 8
#define LL 2048
#define MM (BB * LL)        // 16384 tokens
#define LPAD (LL + 2)       // padded sequence for conv
#define NEXP (2 * DD)       // 2048

typedef __attribute__((ext_vector_type(16))) _Float16 v16h;
typedef __attribute__((ext_vector_type(8)))  float    v8f;
typedef __attribute__((ext_vector_type(4)))  _Float16 v4h;
typedef __attribute__((ext_vector_type(4)))  float    v4f;

#define WMMA_F16(Cacc, Afrag, Bfrag)                                          \
  Cacc = __builtin_amdgcn_wmma_f32_16x16x32_f16(false, Afrag, false, Bfrag,   \
                                                (short)0, Cacc, false, false)

// ---------------- weight conversion ----------------
__global__ void cvt_f32_f16_kernel(const float* __restrict__ s,
                                   _Float16* __restrict__ d, int n) {
  int i = blockIdx.x * 256 + threadIdx.x;
  if (i < n) d[i] = (_Float16)s[i];
}

// conv_w (D,D,3) f32 -> d[k*D*D + o*D + i] f16 (3 contiguous DxD matrices)
__global__ void cvt_convw_kernel(const float* __restrict__ s,
                                 _Float16* __restrict__ d) {
  int i = blockIdx.x * 256 + threadIdx.x;   // i = k*D*D + o*D + ii
  int k = i / (DD * DD);
  int rem = i - k * (DD * DD);
  d[i] = (_Float16)s[(size_t)rem * 3 + k];
}

// ---------------- LayerNorm -> padded f16 activations ----------------
__global__ void __launch_bounds__(256)
ln_kernel(const float* __restrict__ x, const float* __restrict__ gam,
          const float* __restrict__ bet, _Float16* __restrict__ ypad) {
  const int r = blockIdx.x;               // 0 .. B*LPAD-1
  const int b = r / LPAD;
  const int l = r - b * LPAD;
  const int tid = threadIdx.x;
  _Float16* orow = ypad + (size_t)r * DD;
  if (l == 0 || l == LPAD - 1) {          // zero pad rows
    v4h z = {};
    ((v4h*)orow)[tid] = z;
    return;
  }
  const float* xrow = x + (size_t)(b * LL + (l - 1)) * DD;
  v4f xv = ((const v4f*)xrow)[tid];
  float s = xv.x + xv.y + xv.z + xv.w;
  float q = xv.x * xv.x + xv.y * xv.y + xv.z * xv.z + xv.w * xv.w;
  __shared__ float ssum[256];
  __shared__ float ssq[256];
  ssum[tid] = s; ssq[tid] = q;
  __syncthreads();
  for (int off = 128; off > 0; off >>= 1) {
    if (tid < off) { ssum[tid] += ssum[tid + off]; ssq[tid] += ssq[tid + off]; }
    __syncthreads();
  }
  const float mu = ssum[0] * (1.0f / DD);
  const float var = ssq[0] * (1.0f / DD) - mu * mu;
  const float rinv = rsqrtf(var + 1e-5f);
  v4f gv = ((const v4f*)gam)[tid];
  v4f bv = ((const v4f*)bet)[tid];
  v4h o;
  o.x = (_Float16)((xv.x - mu) * rinv * gv.x + bv.x);
  o.y = (_Float16)((xv.y - mu) * rinv * gv.y + bv.y);
  o.z = (_Float16)((xv.z - mu) * rinv * gv.z + bv.z);
  o.w = (_Float16)((xv.w - mu) * rinv * gv.w + bv.w);
  ((v4h*)orow)[tid] = o;
}

// ---------------- WMMA epilogue helpers ----------------
// C/D layout: VGPR v, lanes 0-15: M=v, N=lane; lanes 16-31: M=v+8, N=lane-16
__device__ __forceinline__ void store_acc_f16(const v8f c, _Float16* __restrict__ out,
                                              int ldo, int mBase, int nBase, int lane,
                                              const float* __restrict__ bias) {
  const int n = nBase + (lane & 15);
  const int m0 = mBase + ((lane >> 4) << 3);
  const float bv = bias[n];
#pragma unroll
  for (int v = 0; v < 8; ++v)
    out[(size_t)(m0 + v) * ldo + n] = (_Float16)(c[v] + bv);
}

__device__ __forceinline__ void store_acc_f32_res(const v8f c, float* __restrict__ out,
                                                  int ldo, int mBase, int nBase, int lane,
                                                  const float* __restrict__ bias,
                                                  const float* __restrict__ resid) {
  const int n = nBase + (lane & 15);
  const int m0 = mBase + ((lane >> 4) << 3);
  const float bv = bias[n];
#pragma unroll
  for (int v = 0; v < 8; ++v) {
    size_t idx = (size_t)(m0 + v) * ldo + n;
    out[idx] = c[v] + bv + resid[idx];
  }
}

// ---------------- Conv1d(k=3, pad=1) as 3 accumulated WMMA GEMMs ----------------
// Wave tile 32x64, block tile 128x128 (8 waves: 4 along M x 2 along N).
__global__ void __launch_bounds__(256)
conv_gemm_kernel(const _Float16* __restrict__ ypad, const _Float16* __restrict__ w,
                 const float* __restrict__ bias, _Float16* __restrict__ yc) {
  const int lane = threadIdx.x & 31;
  const int wv = threadIdx.x >> 5;
  const int mTile = blockIdx.x * 128 + (wv & 3) * 32;
  const int nTile = blockIdx.y * 128 + (wv >> 2) * 64;
  const int b = mTile >> 11;                            // 2048 % 128 == 0
  const int rBase = b * LPAD + (mTile & (LL - 1));
  const int lr = lane & 15;
  const int lk = (lane >> 4) << 4;                      // K half: 0 or 16
  const _Float16* ap0 = ypad + (size_t)(rBase + lr) * DD + lk;
  const _Float16* bp0 = w + (size_t)(nTile + lr) * DD + lk;
  v8f acc[2][4] = {};
#pragma unroll
  for (int t = 0; t < 3; ++t) {
    const _Float16* at = ap0 + (size_t)t * DD;          // tap = +t row shift
    const _Float16* bt = bp0 + (size_t)t * DD * DD;
    for (int kk = 0; kk < DD; kk += 32) {
      v16h Af[2], Bf[4];
#pragma unroll
      for (int i = 0; i < 2; ++i)
        Af[i] = *(const v16h*)(at + (size_t)(i * 16) * DD + kk);
#pragma unroll
      for (int j = 0; j < 4; ++j)
        Bf[j] = *(const v16h*)(bt + (size_t)(j * 16) * DD + kk);
      __builtin_prefetch((const void*)(at + kk + 256), 0, 3);
      __builtin_prefetch((const void*)(bt + kk + 256), 0, 3);
#pragma unroll
      for (int i = 0; i < 2; ++i)
#pragma unroll
        for (int j = 0; j < 4; ++j)
          WMMA_F16(acc[i][j], Af[i], Bf[j]);
    }
  }
#pragma unroll
  for (int i = 0; i < 2; ++i)
#pragma unroll
    for (int j = 0; j < 4; ++j)
      store_acc_f16(acc[i][j], yc, DD, mTile + i * 16, nTile + j * 16, lane, bias);
}

// ---------------- Generic WMMA GEMM: out[m,n] = sum_k A[m,k] W[n,k] + bias[n] ----------------
// Wave tile 32x64, block tile 128x128.
template <bool RESID>
__global__ void __launch_bounds__(256)
gemm_kernel(const _Float16* __restrict__ A, const _Float16* __restrict__ W,
            const float* __restrict__ bias, const float* __restrict__ resid,
            _Float16* __restrict__ outH, float* __restrict__ outF,
            const int K, const int N) {
  const int lane = threadIdx.x & 31;
  const int wv = threadIdx.x >> 5;
  const int mTile = blockIdx.x * 128 + (wv & 3) * 32;
  const int nTile = blockIdx.y * 128 + (wv >> 2) * 64;
  const int lr = lane & 15;
  const int lk = (lane >> 4) << 4;
  const _Float16* ap = A + (size_t)(mTile + lr) * K + lk;
  const _Float16* bp = W + (size_t)(nTile + lr) * K + lk;
  v8f acc[2][4] = {};
  for (int kk = 0; kk < K; kk += 32) {
    v16h Af[2], Bf[4];
#pragma unroll
    for (int i = 0; i < 2; ++i)
      Af[i] = *(const v16h*)(ap + (size_t)(i * 16) * K + kk);
#pragma unroll
    for (int j = 0; j < 4; ++j)
      Bf[j] = *(const v16h*)(bp + (size_t)(j * 16) * K + kk);
    __builtin_prefetch((const void*)(ap + kk + 256), 0, 3);
    __builtin_prefetch((const void*)(bp + kk + 256), 0, 3);
#pragma unroll
    for (int i = 0; i < 2; ++i)
#pragma unroll
      for (int j = 0; j < 4; ++j)
        WMMA_F16(acc[i][j], Af[i], Bf[j]);
  }
  if (RESID) {
#pragma unroll
    for (int i = 0; i < 2; ++i)
#pragma unroll
      for (int j = 0; j < 4; ++j)
        store_acc_f32_res(acc[i][j], outF, N, mTile + i * 16, nTile + j * 16,
                          lane, bias, resid);
  } else {
#pragma unroll
    for (int i = 0; i < 2; ++i)
#pragma unroll
      for (int j = 0; j < 4; ++j)
        store_acc_f16(acc[i][j], outH, N, mTile + i * 16, nTile + j * 16, lane, bias);
  }
}

// ---------------- scalar gate: g[tok] = sigmoid(yc[tok,:] . wg + wg_b) ----------------
__global__ void __launch_bounds__(256)
gate_kernel(const _Float16* __restrict__ yc, const float* __restrict__ wg,
            const float* __restrict__ wgb, float* __restrict__ g) {
  const int lane = threadIdx.x & 31;
  const int wv = threadIdx.x >> 5;
  const int tok = blockIdx.x * 8 + wv;
  const _Float16* row = yc + (size_t)tok * DD;
  float s = 0.0f;
  for (int i = lane; i < DD; i += 32) s += (float)row[i] * wg[i];
#pragma unroll
  for (int off = 16; off > 0; off >>= 1) s += __shfl_xor(s, off, 32);
  if (lane == 0) g[tok] = 1.0f / (1.0f + __expf(-(s + wgb[0])));
}

// ---------------- sequential selective-scan: h = (1-g)h + g*x ----------------
// b and the g[] address are derived from blockIdx only -> wave-uniform scalar loads.
__global__ void __launch_bounds__(256)
scan_kernel(const float* __restrict__ g, const _Float16* __restrict__ yc,
            _Float16* __restrict__ hs) {
  const int b = blockIdx.x >> 2;                        // 4 blocks per batch
  const int d = (blockIdx.x & 3) * 256 + threadIdx.x;   // channel
  const float* gb = g + (size_t)b * LL;
  size_t base = (size_t)b * LL * DD + d;
  float h = 0.0f;
  for (int l = 0; l < LL; ++l) {
    const float gt = gb[l];                             // uniform -> s_load
    const float xt = (float)yc[base + (size_t)l * DD];
    h = (1.0f - gt) * h + gt * xt;
    hs[base + (size_t)l * DD] = (_Float16)h;
  }
}

// ---------------- gated MLP activation: gm = sigmoid(fc[:, :D]) * fc[:, D:] ----------------
__global__ void __launch_bounds__(256)
gact_kernel(const _Float16* __restrict__ fc, _Float16* __restrict__ gm) {
  const int gid = blockIdx.x * 256 + threadIdx.x;  // over M*D
  const int m = gid >> 10;
  const int j = gid & (DD - 1);
  const float a = (float)fc[(size_t)m * NEXP + j];
  const float v = (float)fc[(size_t)m * NEXP + DD + j];
  gm[gid] = (_Float16)(v * (1.0f / (1.0f + __expf(-a))));
}

// ---------------- workspace layout (bytes) ----------------
constexpr size_t OFF_WCONV = 0;
constexpr size_t OFF_WOUT  = OFF_WCONV + (size_t)3 * DD * DD * 2;
constexpr size_t OFF_WFC   = OFF_WOUT  + (size_t)DD * DD * 2;
constexpr size_t OFF_WO2   = OFF_WFC   + (size_t)2 * DD * DD * 2;
constexpr size_t OFF_YPAD  = OFF_WO2   + (size_t)DD * DD * 2;
constexpr size_t OFF_YC    = OFF_YPAD  + (size_t)BB * LPAD * DD * 2;
constexpr size_t OFF_G     = OFF_YC    + (size_t)MM * DD * 2;
constexpr size_t OFF_HS    = OFF_G     + (size_t)MM * 4;
constexpr size_t OFF_Y2    = OFF_HS    + (size_t)MM * DD * 2;
constexpr size_t OFF_FC    = OFF_Y2    + (size_t)MM * DD * 2;
constexpr size_t OFF_GM    = OFF_FC    + (size_t)MM * NEXP * 2;

extern "C" void kernel_launch(void* const* d_in, const int* in_sizes, int n_in,
                              void* d_out, int out_size, void* d_ws, size_t ws_size,
                              hipStream_t stream) {
  (void)in_sizes; (void)n_in; (void)out_size; (void)ws_size;
  const float* x      = (const float*)d_in[0];
  const float* ln_g   = (const float*)d_in[1];
  const float* ln_b   = (const float*)d_in[2];
  const float* conv_w = (const float*)d_in[3];
  const float* conv_b = (const float*)d_in[4];
  const float* wg_w   = (const float*)d_in[5];
  const float* wg_b   = (const float*)d_in[6];
  const float* wout_w = (const float*)d_in[7];
  const float* wout_b = (const float*)d_in[8];
  const float* fc_w   = (const float*)d_in[9];
  const float* fc_b   = (const float*)d_in[10];
  const float* out_w  = (const float*)d_in[11];
  const float* out_b  = (const float*)d_in[12];

  char* ws = (char*)d_ws;
  _Float16* wConv16 = (_Float16*)(ws + OFF_WCONV);
  _Float16* wOut16  = (_Float16*)(ws + OFF_WOUT);
  _Float16* wFc16   = (_Float16*)(ws + OFF_WFC);
  _Float16* wO2_16  = (_Float16*)(ws + OFF_WO2);
  _Float16* ypad16  = (_Float16*)(ws + OFF_YPAD);
  _Float16* yc16    = (_Float16*)(ws + OFF_YC);
  float*    gvec    = (float*)(ws + OFF_G);
  _Float16* hs16    = (_Float16*)(ws + OFF_HS);
  _Float16* y2_16   = (_Float16*)(ws + OFF_Y2);
  _Float16* fc16    = (_Float16*)(ws + OFF_FC);
  _Float16* gm16    = (_Float16*)(ws + OFF_GM);

  // weight conversions (deterministic, cheap: ~7M elements)
  cvt_convw_kernel<<<3 * DD * DD / 256, 256, 0, stream>>>(conv_w, wConv16);
  cvt_f32_f16_kernel<<<DD * DD / 256, 256, 0, stream>>>(wout_w, wOut16, DD * DD);
  cvt_f32_f16_kernel<<<2 * DD * DD / 256, 256, 0, stream>>>(fc_w, wFc16, 2 * DD * DD);
  cvt_f32_f16_kernel<<<DD * DD / 256, 256, 0, stream>>>(out_w, wO2_16, DD * DD);

  // 1. LayerNorm -> padded f16 activations
  ln_kernel<<<BB * LPAD, 256, 0, stream>>>(x, ln_g, ln_b, ypad16);

  // 2. Conv1d as 3-tap WMMA GEMM (+bias) -> yc f16
  conv_gemm_kernel<<<dim3(MM / 128, DD / 128), 256, 0, stream>>>(ypad16, wConv16, conv_b, yc16);

  // 3. scalar gate per token
  gate_kernel<<<MM / 8, 256, 0, stream>>>(yc16, wg_w, wg_b, gvec);

  // 4. selective scan over L
  scan_kernel<<<BB * DD / 256, 256, 0, stream>>>(gvec, yc16, hs16);

  // 5. wout projection (WMMA) -> y2 f16
  gemm_kernel<false><<<dim3(MM / 128, DD / 128), 256, 0, stream>>>(
      hs16, wOut16, wout_b, nullptr, y2_16, nullptr, DD, DD);

  // 6. fc projection (WMMA, N=2D) -> fc f16
  gemm_kernel<false><<<dim3(MM / 128, NEXP / 128), 256, 0, stream>>>(
      y2_16, wFc16, fc_b, nullptr, fc16, nullptr, DD, NEXP);

  // 7. gated activation
  gact_kernel<<<MM * DD / 256, 256, 0, stream>>>(fc16, gm16);

  // 8. output projection (WMMA) + bias + residual x -> d_out f32
  gemm_kernel<true><<<dim3(MM / 128, DD / 128), 256, 0, stream>>>(
      gm16, wO2_16, out_b, x, nullptr, (float*)d_out, DD, DD);
}
